// Attention_48765058679198
// MI455X (gfx1250) — matrix operand order repaired and checked by reading, not compiled, not run on hardware
//
#include <hip/hip_runtime.h>

// ---------------------------------------------------------------------------
// Types for CDNA5 WMMA (wave32, 16x16x32 bf16 -> f32)
// ---------------------------------------------------------------------------
typedef __attribute__((ext_vector_type(16))) __bf16 v16bf;
typedef __attribute__((ext_vector_type(8)))  float  v8f;
typedef __attribute__((ext_vector_type(4)))  unsigned int u32x4;

union FragAB { v16bf v; u32x4 q[2]; };

__device__ __forceinline__ v8f wmma_bf16(v16bf a, v16bf b, v8f c) {
  return __builtin_amdgcn_wmma_f32_16x16x32_bf16(
      /*neg_a=*/false, a, /*neg_b=*/false, b,
      /*c_mod=*/(short)0, c, /*reuse_a=*/false, /*reuse_b=*/false);
}

__device__ __forceinline__ unsigned short f2bf(float f) {
  unsigned u = __builtin_bit_cast(unsigned, f);
  unsigned r = u + 0x7FFFu + ((u >> 16) & 1u);   // round-to-nearest-even
  return (unsigned short)(r >> 16);
}

__device__ __forceinline__ unsigned pack2bf(float lo, float hi) {
  return (unsigned)f2bf(lo) | ((unsigned)f2bf(hi) << 16);
}

// CDNA5 async memory->LDS copy (ASYNCcnt-tracked); per-lane 16B.
// VDST operand = LDS byte address (low 32 bits of flat pointer).
__device__ __forceinline__ void async_b128(const unsigned short* gaddr,
                                           unsigned lds_off) {
  asm volatile("global_load_async_to_lds_b128 %0, %1, off"
               :: "v"(lds_off), "v"(gaddr)
               : "memory");
}

__device__ __forceinline__ void wait_async0() {
  asm volatile("s_wait_asynccnt 0x0" ::: "memory");
}

__device__ __forceinline__ unsigned lds_addr(const void* p) {
  return (unsigned)(size_t)p;   // LDS aperture: addr[31:0] is the LDS offset
}

// xor-shuffle within 16-lane rows. v_permlane16_b32 keeps it on VALU
// (one op, immediate lane-select patterns); falls back to ds_bpermute.
__device__ __forceinline__ float xor16(float x, int stage) {
#if __has_builtin(__builtin_amdgcn_permlane16)
  constexpr unsigned SEL0[4] = {0x67452301u, 0x54761032u, 0x32107654u, 0xFEDCBA98u};
  constexpr unsigned SEL1[4] = {0xEFCDAB89u, 0xDCFE98BAu, 0xBA98FEDCu, 0x76543210u};
  unsigned i = __builtin_bit_cast(unsigned, x);
  unsigned r = __builtin_amdgcn_permlane16(i, i, SEL0[stage], SEL1[stage], false, false);
  return __builtin_bit_cast(float, r);
#else
  return __shfl_xor(x, 1 << stage, 32);
#endif
}

// Problem constants
#define BATCH 4
#define SEQ   2048
#define DMODEL 1024
#define NHEADS 16
#define HDIM  64
#define MROWS (BATCH * SEQ)   // 8192

// ---------------------------------------------------------------------------
// Stage 1a: fp32 -> bf16 straight cast (x)
// ---------------------------------------------------------------------------
__global__ __launch_bounds__(256) void cast_f32_bf16(
    const float* __restrict__ src, unsigned short* __restrict__ dst) {
  int i = blockIdx.x * blockDim.x + threadIdx.x;   // handles 4 floats
  float4 v = ((const float4*)src)[i];
  unsigned* d = (unsigned*)dst;
  d[2 * i]     = pack2bf(v.x, v.y);
  d[2 * i + 1] = pack2bf(v.z, v.w);
}

// ---------------------------------------------------------------------------
// Stage 1b: fp32 [rows][cols] -> bf16 [cols][rows] (transpose-cast for weights)
// ---------------------------------------------------------------------------
__global__ __launch_bounds__(256) void transpose_cast(
    const float* __restrict__ src, unsigned short* __restrict__ dst,
    int rows, int cols) {
  __shared__ float tile[32][33];
  int c0 = blockIdx.x * 32, r0 = blockIdx.y * 32;
  int tx = threadIdx.x, ty = threadIdx.y;    // 32 x 8
#pragma unroll
  for (int i = 0; i < 32; i += 8)
    tile[ty + i][tx] = src[(size_t)(r0 + ty + i) * cols + c0 + tx];
  __syncthreads();
#pragma unroll
  for (int i = 0; i < 32; i += 8)
    dst[(size_t)(c0 + ty + i) * rows + r0 + tx] = f2bf(tile[tx][ty + i]);
}

// ---------------------------------------------------------------------------
// Stage 2 & 4: bf16 GEMM  C[M][N] = A[M][K] * B[K][N]  with Bt = [N][K] bf16.
// Block tile 128x128, 8 waves (256 thr), wave tile 32x64 = 2x4 WMMA frags.
// BK=64: 16 WMMAs per barrier. Tile feed: double-buffered
// GLOBAL_LOAD_ASYNC_TO_LDS_B128 (ASYNCcnt), one barrier per K-chunk.
// MODE 0: scatter into Q[b,h,n,d], K[b,h,n,d], V[b,h,d,n]  (N = 3072)
// MODE 1: plain f32 output [M][1024]
// ---------------------------------------------------------------------------
template <int MODE>
__global__ __launch_bounds__(256) void gemm_bf16_wmma(
    const unsigned short* __restrict__ A,
    const unsigned short* __restrict__ Bt,
    float* __restrict__ Cout,
    unsigned short* __restrict__ q_out,
    unsigned short* __restrict__ k_out,
    unsigned short* __restrict__ v_out,
    int Kdim) {
  constexpr int BK = 64;
  __shared__ unsigned short As[2][128][BK + 8];   // +8 bf16 pad vs bank conflicts
  __shared__ unsigned short Bs[2][128][BK + 8];

  const int tid  = threadIdx.x;
  const int lane = tid & 31;
  const int w    = tid >> 5;
  const int wm   = w & 3;        // 4 waves along M
  const int wn   = w >> 2;       // 2 waves along N
  const int hl   = lane >> 4;    // lane half (A/B layout selector)
  const int lm   = lane & 15;
  const int m0   = blockIdx.x * 128;
  const int n0   = blockIdx.y * 128;

  // cooperative async tile fetch: 1024 16B segments each for A and Bt
  auto issue_tile = [&](int p, int k0) {
#pragma unroll
    for (int s = 0; s < 4; ++s) {
      int qq = tid + 256 * s;
      int r  = qq >> 3;
      int cs = (qq & 7) * 8;
      async_b128(A  + (size_t)(m0 + r) * Kdim + k0 + cs, lds_addr(&As[p][r][cs]));
      async_b128(Bt + (size_t)(n0 + r) * Kdim + k0 + cs, lds_addr(&Bs[p][r][cs]));
    }
  };

  v8f acc[2][4];
#pragma unroll
  for (int i = 0; i < 2; ++i)
#pragma unroll
    for (int j = 0; j < 4; ++j)
      acc[i][j] = (v8f){0.f, 0.f, 0.f, 0.f, 0.f, 0.f, 0.f, 0.f};

  issue_tile(0, 0);
  const int nk = Kdim / BK;
  for (int kc = 0; kc < nk; ++kc) {
    const int p = kc & 1;
    wait_async0();        // this wave's share of tile kc landed in LDS
    __syncthreads();      // everyone's share landed; prior reads of buf p^1 done
    if (kc + 1 < nk) issue_tile(p ^ 1, (kc + 1) * BK);

    FragAB a[2][2], b[4][2];
#pragma unroll
    for (int dc = 0; dc < 2; ++dc) {
      const int ao = dc * 32 + (hl ? 8 : 0);   // A: K{0-7,16-23} / {8-15,24-31}
#pragma unroll
      for (int mt = 0; mt < 2; ++mt) {
        int m = wm * 32 + mt * 16 + lm;
        a[mt][dc].q[0] = *(const u32x4*)(&As[p][m][ao]);
        a[mt][dc].q[1] = *(const u32x4*)(&As[p][m][ao + 16]);
      }
      const int bo = dc * 32 + (hl ? 8 : 0);   // B: K{0-7,16-23} / {8-15,24-31}
#pragma unroll
      for (int nt = 0; nt < 4; ++nt) {
        int n = wn * 64 + nt * 16 + lm;
        b[nt][dc].q[0] = *(const u32x4*)(&Bs[p][n][bo]);
        b[nt][dc].q[1] = *(const u32x4*)(&Bs[p][n][bo + 16]);
      }
    }
#pragma unroll
    for (int mt = 0; mt < 2; ++mt)
#pragma unroll
      for (int nt = 0; nt < 4; ++nt)
#pragma unroll
        for (int dc = 0; dc < 2; ++dc)
          acc[mt][nt] = wmma_bf16(a[mt][dc].v, b[nt][dc].v, acc[mt][nt]);
  }

  // epilogue
  if (MODE == 1) {
#pragma unroll
    for (int mt = 0; mt < 2; ++mt)
#pragma unroll
      for (int nt = 0; nt < 4; ++nt)
#pragma unroll
        for (int r = 0; r < 8; ++r) {
          int row = m0 + wm * 32 + mt * 16 + r + 8 * hl;
          int col = n0 + wn * 64 + nt * 16 + lm;
          Cout[(size_t)row * DMODEL + col] = acc[mt][nt][r];
        }
  } else {
    // a 128-wide N-tile never straddles the 1024-aligned q/k/v regions
    const int region = n0 >> 10;   // 0:q 1:k 2:v
    if (region == 2) {             // V: [b,h,d,n] (transposed)
#pragma unroll
      for (int mt = 0; mt < 2; ++mt)
#pragma unroll
        for (int nt = 0; nt < 4; ++nt)
#pragma unroll
          for (int r = 0; r < 8; ++r) {
            int row = m0 + wm * 32 + mt * 16 + r + 8 * hl;
            int col = n0 + wn * 64 + nt * 16 + lm;
            int bb = row >> 11, si = row & (SEQ - 1);
            int cm = col & (DMODEL - 1);
            int hh = cm >> 6, dd = cm & (HDIM - 1);
            v_out[(((size_t)(bb * NHEADS + hh)) * HDIM + dd) * SEQ + si] =
                f2bf(acc[mt][nt][r]);
          }
    } else {                       // Q / K: [b,h,n,d]
      unsigned short* __restrict__ dst = (region == 0) ? q_out : k_out;
#pragma unroll
      for (int mt = 0; mt < 2; ++mt)
#pragma unroll
        for (int nt = 0; nt < 4; ++nt)
#pragma unroll
          for (int r = 0; r < 8; ++r) {
            int row = m0 + wm * 32 + mt * 16 + r + 8 * hl;
            int col = n0 + wn * 64 + nt * 16 + lm;
            int bb = row >> 11, si = row & (SEQ - 1);
            int cm = col & (DMODEL - 1);
            int hh = cm >> 6, dd = cm & (HDIM - 1);
            dst[(((size_t)(bb * NHEADS + hh)) * SEQ + si) * HDIM + dd] =
                f2bf(acc[mt][nt][r]);
          }
    }
  }
}

// ---------------------------------------------------------------------------
// Stage 3: flash attention, causal, online softmax.
// Grid: B*H*(SEQ/128) blocks, 256 threads (8 waves x 16 Q-rows each).
// Q[b,h,n,d] bf16; K[b,h,n,d] bf16 (rows are Bt layout for QK^T);
// V[b,h,d,n] bf16 (rows are Bt layout for P*V). Output Oa[M][D] bf16.
// K/V tiles double-buffered via async-to-LDS; one barrier per key block.
// Mask VALU only on the 4 diagonal-band key blocks (block-uniform split).
// Row max via v_permlane16 (VALU); row sums via WMMA with all-ones B.
// ---------------------------------------------------------------------------
__global__ __launch_bounds__(256) void flash_attn_wmma(
    const unsigned short* __restrict__ Q,
    const unsigned short* __restrict__ Kd,
    const unsigned short* __restrict__ Vt,
    unsigned short* __restrict__ Oa) {
  __shared__ unsigned short Ks[2][32][72];      // [key][d]   (+8 pad)
  __shared__ unsigned short Vs[2][64][40];      // [d][key]   (+8 pad)
  __shared__ unsigned short Ps[8][16][40];      // per-wave P tile [row][key]

  const int tid  = threadIdx.x;
  const int lane = tid & 31;
  const int w    = tid >> 5;
  const int hl   = lane >> 4;
  const int lm   = lane & 15;
  const int rb   = 15 - (blockIdx.x & 15);   // heavy (high-rb) blocks launch first
  const int bh   = blockIdx.x >> 4;          // b*NHEADS + h
  const size_t base = (size_t)bh * SEQ * HDIM;

  // cooperative async K/V tile fetch (one b128 each per thread)
  auto issue_kv = [&](int p, int j0) {
    int kr = tid >> 3, ks = (tid & 7) * 8;
    async_b128(Kd + base + (size_t)(j0 + kr) * HDIM + ks, lds_addr(&Ks[p][kr][ks]));
    int vr = tid >> 2, vs = (tid & 3) * 8;
    async_b128(Vt + base + (size_t)vr * SEQ + j0 + vs, lds_addr(&Vs[p][vr][vs]));
  };

  // Q fragments for this wave's 16 rows (d = 64 -> 2 K-chunks)
  FragAB qf[2];
  {
    const int qrow = rb * 128 + w * 16 + lm;
    const unsigned short* qp = Q + base + (size_t)qrow * HDIM;
    const int aoff = hl ? 8 : 0;
    qf[0].q[0] = *(const u32x4*)(qp + aoff);
    qf[0].q[1] = *(const u32x4*)(qp + aoff + 16);
    qf[1].q[0] = *(const u32x4*)(qp + 32 + aoff);
    qf[1].q[1] = *(const u32x4*)(qp + 32 + aoff + 16);
  }

  // all-ones bf16 B fragment (for row sums of P on the XDL pipe)
  FragAB onesf;
  onesf.q[0] = (u32x4){0x3F803F80u, 0x3F803F80u, 0x3F803F80u, 0x3F803F80u};
  onesf.q[1] = onesf.q[0];

  v8f o[4];
#pragma unroll
  for (int i = 0; i < 4; ++i)
    o[i] = (v8f){0.f, 0.f, 0.f, 0.f, 0.f, 0.f, 0.f, 0.f};
  float mrow[8], lrow[8];
#pragma unroll
  for (int r = 0; r < 8; ++r) { mrow[r] = -3.0e38f; lrow[r] = 0.f; }

  const int rowBase = rb * 128 + w * 16 + 8 * hl;  // row for acc element r = rowBase+r
  const float scale = 0.125f;                       // 64^-0.5
  const int nkb = (rb + 1) * 4;                     // key blocks up to diagonal

  auto process = [&](int jb, bool masked) {
    const int j0 = jb * 32;
    const int p  = jb & 1;
    wait_async0();
    __syncthreads();
    if (jb + 1 < nkb) issue_kv(p ^ 1, (jb + 1) * 32);

    // preload all K/V fragments into distinct regs so DS loads pipeline
    FragAB kf[2][2], vf[4];
#pragma unroll
    for (int kt = 0; kt < 2; ++kt)
#pragma unroll
      for (int dc = 0; dc < 2; ++dc) {
        int krow = kt * 16 + lm;
        int off  = dc * 32 + (hl ? 8 : 0);
        kf[kt][dc].q[0] = *(const u32x4*)(&Ks[p][krow][off]);
        kf[kt][dc].q[1] = *(const u32x4*)(&Ks[p][krow][off + 16]);
      }
#pragma unroll
    for (int nt = 0; nt < 4; ++nt) {
      int vrow = nt * 16 + lm;
      int off  = hl ? 8 : 0;
      vf[nt].q[0] = *(const u32x4*)(&Vs[p][vrow][off]);
      vf[nt].q[1] = *(const u32x4*)(&Vs[p][vrow][off + 16]);
    }

    // S = Q * K^T   (2 key sub-tiles x 2 d-chunks)
    v8f s[2];
    s[0] = (v8f){0.f,0.f,0.f,0.f,0.f,0.f,0.f,0.f};
    s[1] = (v8f){0.f,0.f,0.f,0.f,0.f,0.f,0.f,0.f};
#pragma unroll
    for (int kt = 0; kt < 2; ++kt)
#pragma unroll
      for (int dc = 0; dc < 2; ++dc)
        s[kt] = wmma_bf16(qf[dc].v, kf[kt][dc].v, s[kt]);

    // scale (+ causal mask only in the diagonal band)
#pragma unroll
    for (int kt = 0; kt < 2; ++kt) {
      int col = j0 + kt * 16 + lm;
#pragma unroll
      for (int r = 0; r < 8; ++r) {
        float v = s[kt][r] * scale;
        s[kt][r] = (masked && col > rowBase + r) ? -50000.0f : v;
      }
    }

    // online softmax: row max via permlane16 (stays within 16-lane rows)
    float alpha[8];
#pragma unroll
    for (int r = 0; r < 8; ++r) {
      float mx = fmaxf(s[0][r], s[1][r]);
#pragma unroll
      for (int st = 0; st < 4; ++st)
        mx = fmaxf(mx, xor16(mx, st));
      float mnew = fmaxf(mrow[r], mx);
      alpha[r] = __expf(mrow[r] - mnew);
      mrow[r] = mnew;
      s[0][r] = __expf(s[0][r] - mnew);
      s[1][r] = __expf(s[1][r] - mnew);
    }

    // rescale O accumulators
#pragma unroll
    for (int nt = 0; nt < 4; ++nt)
#pragma unroll
      for (int r = 0; r < 8; ++r)
        o[nt][r] *= alpha[r];

    // P: C layout -> LDS -> A layout (bf16)
#pragma unroll
    for (int kt = 0; kt < 2; ++kt)
#pragma unroll
      for (int r = 0; r < 8; ++r)
        Ps[w][r + 8 * hl][kt * 16 + lm] = f2bf(s[kt][r]);
    // same-wave DS ops are in-order; only this wave touches Ps[w]
    FragAB pf;
    {
      const int aoff = hl ? 8 : 0;
      pf.q[0] = *(const u32x4*)(&Ps[w][lm][aoff]);
      pf.q[1] = *(const u32x4*)(&Ps[w][lm][aoff + 16]);
    }

    // row sums of P via WMMA against all-ones B (replaces 32 shuffle ops)
    v8f rsum = wmma_bf16(pf.v, onesf.v,
                         (v8f){0.f,0.f,0.f,0.f,0.f,0.f,0.f,0.f});
#pragma unroll
    for (int r = 0; r < 8; ++r)
      lrow[r] = lrow[r] * alpha[r] + rsum[r];

    // O += P * V   (4 d-tiles)
#pragma unroll
    for (int nt = 0; nt < 4; ++nt)
      o[nt] = wmma_bf16(pf.v, vf[nt].v, o[nt]);
  };

  issue_kv(0, 0);
  const int nfull = rb * 4;   // key blocks provably unmasked for ALL rows here
  for (int jb = 0; jb < nfull; ++jb) process(jb, false);
  for (int jb = nfull; jb < nkb; ++jb) process(jb, true);

  // normalize + write Oattn[b*SEQ+seq][h*64+d] bf16
  const int b_ = bh >> 4, h_ = bh & (NHEADS - 1);
  float inv[8];
#pragma unroll
  for (int r = 0; r < 8; ++r) inv[r] = 1.0f / lrow[r];
#pragma unroll
  for (int nt = 0; nt < 4; ++nt) {
#pragma unroll
    for (int r = 0; r < 8; ++r) {
      int seq  = rowBase + r;
      int dcol = h_ * HDIM + nt * 16 + lm;
      Oa[((size_t)(b_ * SEQ + seq)) * DMODEL + dcol] = f2bf(o[nt][r] * inv[r]);
    }
  }
}

// ---------------------------------------------------------------------------
// Launch
// ---------------------------------------------------------------------------
extern "C" void kernel_launch(void* const* d_in, const int* in_sizes, int n_in,
                              void* d_out, int out_size, void* d_ws, size_t ws_size,
                              hipStream_t stream) {
  const float* x   = (const float*)d_in[0];
  const float* Wq  = (const float*)d_in[1];
  const float* Wkv = (const float*)d_in[2];
  const float* Wo  = (const float*)d_in[3];
  float* out = (float*)d_out;

  char* ws = (char*)d_ws;
  size_t off = 0;
  auto alloc = [&](size_t bytes) -> void* {
    void* p = ws + off;
    off += (bytes + 255) & ~(size_t)255;
    return p;
  };
  unsigned short* xb  = (unsigned short*)alloc((size_t)MROWS * DMODEL * 2);       // 16.8 MB
  unsigned short* wt  = (unsigned short*)alloc((size_t)3 * DMODEL * DMODEL * 2);  // 6.3 MB ([3072][1024])
  unsigned short* wot = (unsigned short*)alloc((size_t)DMODEL * DMODEL * 2);      // 2.1 MB
  unsigned short* Qb  = (unsigned short*)alloc((size_t)MROWS * DMODEL * 2);
  unsigned short* Kb  = (unsigned short*)alloc((size_t)MROWS * DMODEL * 2);
  unsigned short* Vb  = (unsigned short*)alloc((size_t)MROWS * DMODEL * 2);
  unsigned short* Oa  = (unsigned short*)alloc((size_t)MROWS * DMODEL * 2);

  // 1. casts / weight transposes
  cast_f32_bf16<<<(MROWS * DMODEL) / (4 * 256), 256, 0, stream>>>(x, xb);
  transpose_cast<<<dim3(DMODEL / 32, DMODEL / 32), dim3(32, 8), 0, stream>>>(
      Wq, wt, DMODEL, DMODEL);
  transpose_cast<<<dim3(2 * DMODEL / 32, DMODEL / 32), dim3(32, 8), 0, stream>>>(
      Wkv, wt + (size_t)DMODEL * DMODEL, DMODEL, 2 * DMODEL);
  transpose_cast<<<dim3(DMODEL / 32, DMODEL / 32), dim3(32, 8), 0, stream>>>(
      Wo, wot, DMODEL, DMODEL);

  // 2. fused qkv projection (N = 3072), scatter to heads (V transposed)
  gemm_bf16_wmma<0><<<dim3(MROWS / 128, 3 * DMODEL / 128), 256, 0, stream>>>(
      xb, wt, nullptr, Qb, Kb, Vb, DMODEL);

  // 3. causal flash attention
  flash_attn_wmma<<<BATCH * NHEADS * (SEQ / 128), 256, 0, stream>>>(Qb, Kb, Vb, Oa);

  // 4. output projection -> fp32
  gemm_bf16_wmma<1><<<dim3(MROWS / 128, DMODEL / 128), 256, 0, stream>>>(
      Oa, wot, out, nullptr, nullptr, nullptr, DMODEL);
}
